// SimpleAttention_46866683134542
// MI455X (gfx1250) — matrix operand order repaired and checked by reading, not compiled, not run on hardware
//
#include <hip/hip_runtime.h>

#define BATCH   2
#define SEQ     2048
#define HIDDEN  1024
#define NHEADS  16
#define NKV     4
#define HDIM    64
#define THETA   100000.0f

typedef __attribute__((ext_vector_type(16))) __bf16 v16bf;
typedef __attribute__((ext_vector_type(8)))  float  v8f;

union Frag16 {
    v16bf          v;
    unsigned short u[16];
    unsigned int   d[8];
};

__device__ __forceinline__ unsigned short f2bf(float x) {
    unsigned int u = __float_as_uint(x);
    u += 0x7FFFu + ((u >> 16) & 1u);     // round-to-nearest-even
    return (unsigned short)(u >> 16);
}

// ---------------------------------------------------------------------------
// fp32 -> bf16, optionally applying the reference RoPE (layout preserved):
//   out[d] = x[d]*cos(t*inv[d%32]) + rot[d]*sin(t*inv[d%32])
//   rot[2i] = -x[2i+1], rot[2i+1] = x[2i], inv[i] = theta^(-2i/64)
// `in` layout: (B*S, nheads, 64) contiguous.
// ---------------------------------------------------------------------------
__global__ void convert_rope_kernel(const float* __restrict__ in,
                                    unsigned short* __restrict__ out,
                                    int nheads, int apply_rope, long long total) {
    long long idx = (long long)blockIdx.x * blockDim.x + threadIdx.x;
    if (idx >= total) return;
    float x = in[idx];
    float o = x;
    if (apply_rope) {
        int d = (int)(idx & 63);
        long long bs = (idx >> 6) / nheads;
        int t = (int)(bs % SEQ);
        int fi = d & 31;
        float inv = __powf(THETA, -(float)(2 * fi) / 64.0f);
        float ang = (float)t * inv;
        float c, s;
        __sincosf(ang, &c, &s);
        float rot = (d & 1) ? in[idx - 1] : -in[idx + 1];
        o = x * c + rot * s;
    }
    out[idx] = f2bf(o);
}

// fp32 W[K,N] -> bf16 Wt[N,K] (transposed so WMMA B-fragments are contiguous)
__global__ void convert_transpose_w_kernel(const float* __restrict__ in,
                                           unsigned short* __restrict__ out,
                                           int K, int N) {
    long long idx = (long long)blockIdx.x * blockDim.x + threadIdx.x;
    long long total = (long long)K * N;
    if (idx >= total) return;
    int k = (int)(idx / N);
    int n = (int)(idx % N);
    out[(size_t)n * K + k] = f2bf(in[idx]);
}

// fp32 V (b,s,kvh,d) -> bf16 Vt (b,kvh,d,s) so P@V B-fragments are contiguous
__global__ void convert_transpose_v_kernel(const float* __restrict__ in,
                                           unsigned short* __restrict__ out) {
    long long idx = (long long)blockIdx.x * blockDim.x + threadIdx.x;
    long long total = (long long)BATCH * SEQ * NKV * HDIM;
    if (idx >= total) return;
    int d = (int)(idx & 63);
    long long r = idx >> 6;
    int hv = (int)(r % NKV);
    long long bs = r / NKV;
    int s = (int)(bs % SEQ);
    int b = (int)(bs / SEQ);
    out[(((size_t)b * NKV + hv) * HDIM + d) * SEQ + s] = f2bf(in[idx]);
}

// ---------------------------------------------------------------------------
// C[M,N] = A[M,K](bf16, row-major) @ Wt[N,K](bf16, pre-transposed) + bias[N].
// Block = 4 waves. Double-buffered: per K-step of 32 the block
//   * waits for the current 64x32 B tile (staged via gfx1250 async
//     global->LDS b128 copies, ASYNCcnt-tracked),
//   * kicks off the async copies for the NEXT K-step (overlapped with math),
//   * each wave computes a 32x64 strip: 2 A-frags (global b128) x 4 B-frags
//     (LDS b128) -> 8 WMMAs.
// LDS rows padded to 40 shorts (80 B) to avoid bank conflicts.
// ---------------------------------------------------------------------------
#define BT_PITCH 40
__global__ __launch_bounds__(128) void gemm_bf16_kernel(
    const unsigned short* __restrict__ A,
    const unsigned short* __restrict__ Bt,
    const float* __restrict__ bias,
    float* __restrict__ C,
    int M, int N, int K)
{
    __shared__ unsigned short Btile[2][64 * BT_PITCH];

    const int tid  = threadIdx.x;
    const int lane = tid & 31;
    const int wave = tid >> 5;
    const int m0 = blockIdx.y * 128 + wave * 32;   // this wave: rows m0..m0+31
    const int n0 = blockIdx.x * 64;
    const int mrow = lane & 15;
    const int half = lane >> 4;

    v8f z = {0.f,0.f,0.f,0.f,0.f,0.f,0.f,0.f};
    v8f acc[2][4] = {{z, z, z, z}, {z, z, z, z}};

    const unsigned short* arow0 = A + (size_t)(m0 + mrow) * K;
    const unsigned short* arow1 = A + (size_t)(m0 + 16 + mrow) * K;

    // this thread's two 16-byte staging transfers (256 total per K-step)
    const int row0 = tid >> 2,          seg0 = tid & 3;
    const int row1 = (tid + 128) >> 2,  seg1 = tid & 3;
    // real LDS addresses (ptrtoint escapes Btile into the asm -> loads kept)
    const unsigned ldsA0 = (unsigned)(uintptr_t)&Btile[0][row0 * BT_PITCH + seg0 * 8];
    const unsigned ldsA1 = (unsigned)(uintptr_t)&Btile[0][row1 * BT_PITCH + seg1 * 8];
    const unsigned ldsB0 = (unsigned)(uintptr_t)&Btile[1][row0 * BT_PITCH + seg0 * 8];
    const unsigned ldsB1 = (unsigned)(uintptr_t)&Btile[1][row1 * BT_PITCH + seg1 * 8];
    const unsigned short* gB0 = Bt + (size_t)(n0 + row0) * K + seg0 * 8;
    const unsigned short* gB1 = Bt + (size_t)(n0 + row1) * K + seg1 * 8;

    // prologue: stage tile for kk = 0 into buffer 0
    {
        unsigned long long ga0 = (unsigned long long)(uintptr_t)gB0;
        unsigned long long ga1 = (unsigned long long)(uintptr_t)gB1;
        asm volatile("global_load_async_to_lds_b128 %0, %1, off"
                     :: "v"(ldsA0), "v"(ga0) : "memory");
        asm volatile("global_load_async_to_lds_b128 %0, %1, off"
                     :: "v"(ldsA1), "v"(ga1) : "memory");
    }

    int ibuf = 0;
    for (int kk = 0; kk < K; kk += 32) {
        // current tile's async copies complete (per wave), then block-wide sync:
        // also guarantees every wave consumed the buffer we are about to refill.
        asm volatile("s_wait_asynccnt 0x0" ::: "memory");
        __syncthreads();

        if (kk + 32 < K) {   // overlap next tile's copies with this step's math
            unsigned l0 = ibuf ? ldsA0 : ldsB0;
            unsigned l1 = ibuf ? ldsA1 : ldsB1;
            unsigned long long ga0 = (unsigned long long)(uintptr_t)(gB0 + kk + 32);
            unsigned long long ga1 = (unsigned long long)(uintptr_t)(gB1 + kk + 32);
            asm volatile("global_load_async_to_lds_b128 %0, %1, off"
                         :: "v"(l0), "v"(ga0) : "memory");
            asm volatile("global_load_async_to_lds_b128 %0, %1, off"
                         :: "v"(l1), "v"(ga1) : "memory");
        }

        // A fragments from global: VGPR v holds K pair (2v + 16*half)
        Frag16 a[2];
        #pragma unroll
        for (int v = 0; v < 8; ++v) {
            int kbase = kk + 2 * v + 16 * half;
            a[0].d[v] = *(const unsigned int*)(arow0 + kbase);
            a[1].d[v] = *(const unsigned int*)(arow1 + kbase);
        }
        const unsigned short* btb = &Btile[ibuf][0];
        #pragma unroll
        for (int nc = 0; nc < 4; ++nc) {
            // B fragment from LDS: element e -> (k = kk + e + 16*half, n = n0+nc*16+mrow)
            Frag16 bf;
            const unsigned int* bp = (const unsigned int*)
                (btb + (nc * 16 + mrow) * BT_PITCH + 16 * half);
            #pragma unroll
            for (int v = 0; v < 8; ++v) bf.d[v] = bp[v];
            acc[0][nc] = __builtin_amdgcn_wmma_f32_16x16x32_bf16(
                false, a[0].v, false, bf.v, (short)0, acc[0][nc], false, false);
            acc[1][nc] = __builtin_amdgcn_wmma_f32_16x16x32_bf16(
                false, a[1].v, false, bf.v, (short)0, acc[1][nc], false, false);
        }
        ibuf ^= 1;
    }
    // C/D layout: VGPR r -> row r + 8*half, col = lane%16
    #pragma unroll
    for (int mt = 0; mt < 2; ++mt) {
        #pragma unroll
        for (int nc = 0; nc < 4; ++nc) {
            int col = n0 + nc * 16 + mrow;
            float bb = bias[col];
            #pragma unroll
            for (int r = 0; r < 8; ++r) {
                C[(size_t)(m0 + mt * 16 + r + 8 * half) * N + col] = acc[mt][nc][r] + bb;
            }
        }
    }
}

// ---------------------------------------------------------------------------
// Flash attention: one wave per (batch, head, 16-row query tile).
// Q: (B*S, NHEADS, 64) bf16; K: (B*S, NKV, 64) bf16; Vt: (B, NKV, 64, S) bf16.
// Online softmax; 32 keys/iter (2 score WMMA-chunks -> P(16x32) -> 4 PV WMMAs).
// ---------------------------------------------------------------------------
__global__ __launch_bounds__(32) void flash_attn_kernel(
    const unsigned short* __restrict__ Qb,
    const unsigned short* __restrict__ Kb,
    const unsigned short* __restrict__ Vt,
    const int* __restrict__ amask,
    unsigned short* __restrict__ Ctx)
{
    __shared__ unsigned short Pst[16 * 32];   // P tile, row-major, for C->A transpose

    const int lane = threadIdx.x & 31;
    const int mrow = lane & 15;
    const int half = lane >> 4;
    const int q0  = blockIdx.x * 16;
    const int h   = blockIdx.y;
    const int b   = blockIdx.z;
    const int kvh = h >> 2;                   // NHEADS/NKV = 4

    // Q A-fragments (K-dim 64 -> two 16x32 fragments)
    Frag16 qa[2];
    const unsigned short* qrow = Qb + ((size_t)(b * SEQ + q0 + mrow) * NHEADS + h) * HDIM;
    #pragma unroll
    for (int kk = 0; kk < 2; ++kk) {
        #pragma unroll
        for (int v = 0; v < 8; ++v) {
            int kbase = 32 * kk + 2 * v + 16 * half;
            qa[kk].d[v] = *(const unsigned int*)(qrow + kbase);
        }
    }

    v8f z = {0.f,0.f,0.f,0.f,0.f,0.f,0.f,0.f};
    v8f o[4] = {z, z, z, z};
    float mrun[8], lrun[8];
    #pragma unroll
    for (int r = 0; r < 8; ++r) { mrun[r] = -1e30f; lrun[r] = 0.f; }

    const float scale = 0.125f;               // 1/sqrt(64)
    const int kend = q0 + 16;                 // causal horizon for this tile

    for (int kb = 0; kb < kend; kb += 32) {
        float p[2][8];
        float rmax[8];
        #pragma unroll
        for (int r = 0; r < 8; ++r) rmax[r] = -1e30f;

        #pragma unroll
        for (int c = 0; c < 2; ++c) {
            const int keyb = kb + c * 16;
            // B fragment of K^T: lane key = keyb + mrow; dims contiguous per half
            const unsigned short* krow =
                Kb + ((size_t)(b * SEQ + keyb + mrow) * NKV + kvh) * HDIM;
            v8f sc = z;
            #pragma unroll
            for (int kk = 0; kk < 2; ++kk) {
                Frag16 bk;
                const unsigned int* kp = (const unsigned int*)(krow + kk * 32 + 16 * half);
                #pragma unroll
                for (int v = 0; v < 8; ++v) bk.d[v] = kp[v];
                sc = __builtin_amdgcn_wmma_f32_16x16x32_bf16(
                    false, qa[kk].v, false, bk.v, (short)0, sc, false, false);
            }
            const int key = keyb + mrow;      // this lane's key column
            const int kvalid = amask[b * SEQ + key];
            #pragma unroll
            for (int r = 0; r < 8; ++r) {
                int qi = q0 + r + 8 * half;
                float vsc = sc[r] * scale;
                vsc = (key > qi || !kvalid) ? -1e30f : vsc;
                p[c][r] = vsc;
                rmax[r] = fmaxf(rmax[r], vsc);
            }
        }

        // per-row max over the 16 lanes sharing each row (xor masks stay in half)
        #pragma unroll
        for (int r = 0; r < 8; ++r) {
            float v = rmax[r];
            v = fmaxf(v, __shfl_xor(v, 1, 32));
            v = fmaxf(v, __shfl_xor(v, 2, 32));
            v = fmaxf(v, __shfl_xor(v, 4, 32));
            v = fmaxf(v, __shfl_xor(v, 8, 32));
            rmax[r] = v;
        }

        float alpha[8];
        #pragma unroll
        for (int r = 0; r < 8; ++r) {
            float nm = fmaxf(mrun[r], rmax[r]);
            alpha[r] = __expf(mrun[r] - nm);
            mrun[r] = nm;
            float psum = 0.f;
            #pragma unroll
            for (int c = 0; c < 2; ++c) {
                float e = __expf(p[c][r] - nm);
                p[c][r] = e;
                psum += e;
            }
            psum += __shfl_xor(psum, 1, 32);
            psum += __shfl_xor(psum, 2, 32);
            psum += __shfl_xor(psum, 4, 32);
            psum += __shfl_xor(psum, 8, 32);
            lrun[r] = lrun[r] * alpha[r] + psum;
        }

        // rescale running O
        #pragma unroll
        for (int nc = 0; nc < 4; ++nc)
            #pragma unroll
            for (int r = 0; r < 8; ++r)
                o[nc][r] *= alpha[r];

        // C-layout -> A-layout transpose of P through LDS
        __syncthreads();
        #pragma unroll
        for (int c = 0; c < 2; ++c)
            #pragma unroll
            for (int r = 0; r < 8; ++r)
                Pst[(r + 8 * half) * 32 + c * 16 + mrow] = f2bf(p[c][r]);
        __syncthreads();

        Frag16 pa;
        #pragma unroll
        for (int v = 0; v < 8; ++v) {
            int kbase = 2 * v + 16 * half;
            pa.d[v] = *(const unsigned int*)(&Pst[mrow * 32 + kbase]);
        }

        // P(16x32) @ V(32x64): 4 WMMAs; Vt gives 16 contiguous keys per lane
        #pragma unroll
        for (int nc = 0; nc < 4; ++nc) {
            Frag16 bv;
            const unsigned int* vp = (const unsigned int*)
                (Vt + (((size_t)b * NKV + kvh) * HDIM + nc * 16 + mrow) * SEQ
                    + kb + 16 * half);
            #pragma unroll
            for (int v = 0; v < 8; ++v) bv.d[v] = vp[v];
            o[nc] = __builtin_amdgcn_wmma_f32_16x16x32_bf16(
                false, pa.v, false, bv.v, (short)0, o[nc], false, false);
        }
    }

    // normalize and store context as bf16 (b, s, h, d)
    #pragma unroll
    for (int nc = 0; nc < 4; ++nc) {
        #pragma unroll
        for (int r = 0; r < 8; ++r) {
            float denom = lrun[r];
            float val = (denom > 0.f) ? o[nc][r] / denom : 0.f;
            Ctx[((size_t)(b * SEQ + q0 + r + 8 * half) * NHEADS + h) * HDIM + nc * 16 + mrow]
                = f2bf(val);
        }
    }
}

// ---------------------------------------------------------------------------
extern "C" void kernel_launch(void* const* d_in, const int* in_sizes, int n_in,
                              void* d_out, int out_size, void* d_ws, size_t ws_size,
                              hipStream_t stream) {
    (void)in_sizes; (void)n_in; (void)out_size; (void)ws_size;
    const float* X   = (const float*)d_in[0];
    const int*   am  = (const int*)d_in[1];
    const float* Wq  = (const float*)d_in[2];
    const float* bq  = (const float*)d_in[3];
    const float* Wk  = (const float*)d_in[4];
    const float* bk  = (const float*)d_in[5];
    const float* Wv  = (const float*)d_in[6];
    const float* bv  = (const float*)d_in[7];
    const float* Wo  = (const float*)d_in[8];
    const float* bo  = (const float*)d_in[9];
    float* out = (float*)d_out;

    const size_t M = (size_t)BATCH * SEQ;          // 4096
    const size_t KVD = (size_t)NKV * HDIM;         // 256

    char* ws = (char*)d_ws;
    size_t off = 0;
    auto alloc = [&](size_t bytes) -> char* {
        char* p = ws + off;
        off += (bytes + 255) & ~(size_t)255;
        return p;
    };
    unsigned short* Xbf  = (unsigned short*)alloc(M * HIDDEN * 2);
    unsigned short* Wqt  = (unsigned short*)alloc((size_t)HIDDEN * HIDDEN * 2);
    unsigned short* Wkt  = (unsigned short*)alloc((size_t)HIDDEN * KVD * 2);
    unsigned short* Wvt  = (unsigned short*)alloc((size_t)HIDDEN * KVD * 2);
    unsigned short* Wot  = (unsigned short*)alloc((size_t)HIDDEN * HIDDEN * 2);
    float*          Qf   = (float*)alloc(M * HIDDEN * 4);
    float*          Kf   = (float*)alloc(M * KVD * 4);
    float*          Vf   = (float*)alloc(M * KVD * 4);
    unsigned short* Qbf  = (unsigned short*)alloc(M * HIDDEN * 2);
    unsigned short* Kbf  = (unsigned short*)alloc(M * KVD * 2);
    unsigned short* Vtb  = (unsigned short*)alloc(M * KVD * 2);
    unsigned short* Ctx  = (unsigned short*)alloc(M * HIDDEN * 2);

    auto cvt = [&](const float* src, unsigned short* dst, long long n,
                   int nheads, int rope) {
        unsigned blocks = (unsigned)((n + 255) / 256);
        convert_rope_kernel<<<blocks, 256, 0, stream>>>(src, dst, nheads, rope, n);
    };
    auto cvtW = [&](const float* src, unsigned short* dst, int K, int N) {
        long long n = (long long)K * N;
        unsigned blocks = (unsigned)((n + 255) / 256);
        convert_transpose_w_kernel<<<blocks, 256, 0, stream>>>(src, dst, K, N);
    };

    // 1. bf16 conversions: activations (row-major), weights (transposed)
    cvt(X, Xbf, (long long)M * HIDDEN, 1, 0);
    cvtW(Wq, Wqt, HIDDEN, HIDDEN);
    cvtW(Wk, Wkt, HIDDEN, (int)KVD);
    cvtW(Wv, Wvt, HIDDEN, (int)KVD);
    cvtW(Wo, Wot, HIDDEN, HIDDEN);

    // 2. QKV projections (WMMA GEMMs, fp32 out)
    gemm_bf16_kernel<<<dim3(HIDDEN / 64, (unsigned)(M / 128)), 128, 0, stream>>>(
        Xbf, Wqt, bq, Qf, (int)M, HIDDEN, HIDDEN);
    gemm_bf16_kernel<<<dim3((unsigned)(KVD / 64), (unsigned)(M / 128)), 128, 0, stream>>>(
        Xbf, Wkt, bk, Kf, (int)M, (int)KVD, HIDDEN);
    gemm_bf16_kernel<<<dim3((unsigned)(KVD / 64), (unsigned)(M / 128)), 128, 0, stream>>>(
        Xbf, Wvt, bv, Vf, (int)M, (int)KVD, HIDDEN);

    // 3. RoPE(Q,K) -> bf16; V -> bf16 transposed (b, kvh, d, s)
    cvt(Qf, Qbf, (long long)M * HIDDEN, NHEADS, 1);
    cvt(Kf, Kbf, (long long)M * KVD,    NKV,    1);
    {
        long long n = (long long)M * KVD;
        unsigned blocks = (unsigned)((n + 255) / 256);
        convert_transpose_v_kernel<<<blocks, 256, 0, stream>>>(Vf, Vtb);
    }

    // 4. flash attention -> bf16 context (b, s, h, d)
    flash_attn_kernel<<<dim3(SEQ / 16, NHEADS, BATCH), 32, 0, stream>>>(
        Qbf, Kbf, Vtb, am, Ctx);

    // 5. output projection -> fp32 d_out
    gemm_bf16_kernel<<<dim3(HIDDEN / 64, (unsigned)(M / 128)), 128, 0, stream>>>(
        Ctx, Wot, bo, out, (int)M, HIDDEN, HIDDEN);
}
